// VPINN_62663572849249
// MI455X (gfx1250) — matrix operand order repaired and checked by reading, not compiled, not run on hardware
//
#include <hip/hip_runtime.h>
#include <hip/hip_bf16.h>

typedef __attribute__((ext_vector_type(16))) _Float16 v16h;
typedef __attribute__((ext_vector_type(8)))  float    v8f;

#define WAVES_PER_BLOCK 8
#define TRIS_PER_WAVE   16
#define TRIS_PER_BLOCK  (WAVES_PER_BLOCK * TRIS_PER_WAVE)

__global__ __launch_bounds__(256) void vpinn_zero(float* __restrict__ assembled, int n,
                                                  float* __restrict__ out) {
  int i = blockIdx.x * blockDim.x + threadIdx.x;
  if (i < n) assembled[i] = 0.0f;
  if (i == 0) out[0] = 0.0f;
}

// One wave handles 16 triangles. One logical GEMM tile:
//   S[16 tris, 12 combos] = A[16, 32] x B[32, 16]
//   A[m, k=2q+a'] = grad_q[tri, q, a']            (row = 32 contiguous floats)
//   B[k=2q+a', j=a*6+b*3+n] = (a'==a) * w[q] * grad_test[b][n][q]   (cols 12..15 = 0)
// computed with 3 chained v_wmma_f32_16x16x32_f16 (hi/lo split -> ~f32 accuracy).
__global__ __launch_bounds__(256) void vpinn_main(
    const float* __restrict__ vertices,
    const int*   __restrict__ triangles,
    const float* __restrict__ grad_q,
    const float* __restrict__ w_quad,
    const float* __restrict__ grad_test,
    float*       __restrict__ assembled,
    int n_tri)
{
  __shared__ float sS[WAVES_PER_BLOCK][16][17];   // padded: stride 17 -> conflict-free

  const int lane    = threadIdx.x & 31;
  const int wave    = threadIdx.x >> 5;
  const int colj    = lane & 15;
  const int hi_half = (lane >> 4) & 1;            // 0: lanes 0-15, 1: lanes 16-31
  const int tri_base = (blockIdx.x * WAVES_PER_BLOCK + wave) * TRIS_PER_WAVE;

  // ---- constant B fragment (per-lane column) ----
  // B layout (16-bit, 32x16): lane n<16 holds col n, K=0..7 & 16..23;
  // lane 16+n holds col n, K=8..15 & 24..31.
  float bc[16];
  if (colj < 12) {
    const int a  = colj / 6;
    const int r  = colj % 6;
    const int b  = r / 3;
    const int nb = r % 3;
#pragma unroll
    for (int i = 0; i < 16; ++i) {
      const int k  = (i < 8) ? (hi_half * 8 + i) : (16 + hi_half * 8 + (i - 8));
      const int q  = k >> 1;
      const int ap = k & 1;
      bc[i] = (ap == a) ? (w_quad[q] * grad_test[b * 96 + nb * 16 + q]) : 0.0f;
    }
  } else {
#pragma unroll
    for (int i = 0; i < 16; ++i) bc[i] = 0.0f;
  }
  v16h bhi, blo;
#pragma unroll
  for (int i = 0; i < 16; ++i) {
    _Float16 h = (_Float16)bc[i];
    bhi[i] = h;
    blo[i] = (_Float16)(bc[i] - (float)h);
  }

  // ---- A fragment: stream grad_q rows ----
  // A layout (16-bit, 16x32): lane m<16 -> row m, K=0..7 & 16..23;
  // lane 16+m -> row m, K=8..15 & 24..31.  Row = 32 contiguous f32.
  const int   m    = tri_base + colj;
  const int   mrow = (m < n_tri) ? m : (n_tri - 1);
  const float keep = (m < n_tri) ? 1.0f : 0.0f;
  const float4* p  = (const float4*)(grad_q + (size_t)mrow * 32 + (hi_half ? 8 : 0));
  float4 c0 = p[0];
  float4 c1 = p[1];
  float4 c2 = p[4];
  float4 c3 = p[5];
  float av[16] = {c0.x, c0.y, c0.z, c0.w, c1.x, c1.y, c1.z, c1.w,
                  c2.x, c2.y, c2.z, c2.w, c3.x, c3.y, c3.z, c3.w};
  v16h ahi, alo;
#pragma unroll
  for (int i = 0; i < 16; ++i) {
    float f = av[i] * keep;
    _Float16 h = (_Float16)f;
    ahi[i] = h;
    alo[i] = (_Float16)(f - (float)h);
  }

  // ---- 3 chained WMMAs: f32-accurate product via hi/lo split ----
  v8f acc = {};
  acc = __builtin_amdgcn_wmma_f32_16x16x32_f16(false, ahi, false, bhi, (short)0, acc, false, false);
  acc = __builtin_amdgcn_wmma_f32_16x16x32_f16(false, alo, false, bhi, (short)0, acc, false, false);
  acc = __builtin_amdgcn_wmma_f32_16x16x32_f16(false, ahi, false, blo, (short)0, acc, false, false);

  // D layout: lane<16 -> D[M=v][N=lane], lane>=16 -> D[M=8+v][N=lane-16]
  const int mb = hi_half * 8;
#pragma unroll
  for (int v = 0; v < 8; ++v) sS[wave][mb + v][colj] = acc[v];

  __syncthreads();

  // ---- per-triangle finalize: lanes 0..15 each own one triangle ----
  if (lane < 16) {
    const int tri = tri_base + lane;
    if (tri < n_tri) {
      const int i0 = triangles[tri * 3 + 0];
      const int i1 = triangles[tri * 3 + 1];
      const int i2 = triangles[tri * 3 + 2];
      const float x0 = vertices[2 * i0], y0 = vertices[2 * i0 + 1];
      const float x1 = vertices[2 * i1], y1 = vertices[2 * i1 + 1];
      const float x2 = vertices[2 * i2], y2 = vertices[2 * i2 + 1];
      const float e1x = x1 - x0, e1y = y1 - y0;
      const float e2x = x2 - x0, e2y = y2 - y0;
      const float det = e1x * e2y - e2x * e1y;
      const float J   = fabsf(det);
      const float inv = 1.0f / det;
      const float BD00 =  e2y * inv;     // B_D[0][0]
      const float BD01 = -e1y * inv;     // B_D[0][1]
      const float BD10 = -e2x * inv;     // B_D[1][0]
      const float BD11 =  e1x * inv;     // B_D[1][1]
      const float* S = &sS[wave][lane][0];
      const int idx[3] = {i0, i1, i2};
#pragma unroll
      for (int n = 0; n < 3; ++n) {
        float val = J * (BD00 * S[n] + BD01 * S[3 + n] + BD10 * S[6 + n] + BD11 * S[9 + n]);
        atomicAdd(&assembled[idx[n]], val);
      }
    }
  }
}

__global__ __launch_bounds__(256) void vpinn_loss(const float* __restrict__ assembled,
                                                  const float* __restrict__ F,
                                                  float* __restrict__ out, int n) {
  __shared__ float red[256];
  float s = 0.0f;
  for (int i = blockIdx.x * blockDim.x + threadIdx.x; i < n; i += gridDim.x * blockDim.x) {
    float d = assembled[i] - F[i];
    s += d * d;
  }
  red[threadIdx.x] = s;
  __syncthreads();
  for (int off = 128; off > 0; off >>= 1) {
    if ((int)threadIdx.x < off) red[threadIdx.x] += red[threadIdx.x + off];
    __syncthreads();
  }
  if (threadIdx.x == 0) atomicAdd(out, red[0]);
}

extern "C" void kernel_launch(void* const* d_in, const int* in_sizes, int n_in,
                              void* d_out, int out_size, void* d_ws, size_t ws_size,
                              hipStream_t stream) {
  const float* vertices   = (const float*)d_in[0];
  const int*   triangles  = (const int*)d_in[1];
  const float* grad_q     = (const float*)d_in[2];
  const float* w_quad     = (const float*)d_in[3];
  const float* grad_test  = (const float*)d_in[4];
  const float* F_vertices = (const float*)d_in[5];
  const int n_vert = in_sizes[0] / 2;
  const int n_tri  = in_sizes[1] / 3;

  float* assembled = (float*)d_ws;     // n_vert floats of scratch
  float* out       = (float*)d_out;    // scalar loss

  const int zb = (n_vert + 255) / 256;
  vpinn_zero<<<zb, 256, 0, stream>>>(assembled, n_vert, out);

  const int tiles  = (n_tri + TRIS_PER_WAVE - 1) / TRIS_PER_WAVE;
  const int blocks = (tiles + WAVES_PER_BLOCK - 1) / WAVES_PER_BLOCK;
  vpinn_main<<<blocks, 256, 0, stream>>>(vertices, triangles, grad_q, w_quad,
                                         grad_test, assembled, n_tri);

  int lb = (n_vert + 255) / 256;
  if (lb > 1024) lb = 1024;
  vpinn_loss<<<lb, 256, 0, stream>>>(assembled, F_vertices, out, n_vert);
}
